// MultiscaleVectorQuantize_18992345382929
// MI455X (gfx1250) — matrix operand order repaired and checked by reading, not compiled, not run on hardware
//
#include <hip/hip_runtime.h>

// ---------------------------------------------------------------------------
// Problem constants (from reference)
// ---------------------------------------------------------------------------
#define BQ      16
#define D_IN    1024
#define TT      4096
#define CB_SIZE 8192
#define CB_DIM  256
#define SC      1024          // scale
#define RATIO   4             // TT / SC

typedef __bf16 v16bf __attribute__((ext_vector_type(16)));
typedef __bf16 v8bf  __attribute__((ext_vector_type(8)));
typedef float  v8f   __attribute__((ext_vector_type(8)));

static __device__ __forceinline__ __bf16 f2bf(float f) { return (__bf16)f; }

static __device__ __forceinline__ v16bf cat16(v8bf lo, v8bf hi) {
  return __builtin_shufflevector(lo, hi, 0,1,2,3,4,5,6,7,8,9,10,11,12,13,14,15);
}

// ---------------------------------------------------------------------------
// 0) zero small accumulator buffer
// ---------------------------------------------------------------------------
__global__ void msvq_zero_kernel(float* p, int n) {
  int i = blockIdx.x * blockDim.x + threadIdx.x;
  if (i < n) p[i] = 0.0f;
}

// ---------------------------------------------------------------------------
// 1) row-wise L2 normalize (+ optional gain) -> bf16   (weight norm / l2norm)
// ---------------------------------------------------------------------------
__global__ __launch_bounds__(256) void msvq_rownorm_kernel(
    const float* __restrict__ V, const float* __restrict__ G,
    __bf16* __restrict__ W, int cols) {
  __shared__ float red[256];
  const int r = blockIdx.x;
  const float* v = V + (long long)r * cols;
  float s = 0.f;
  for (int c = threadIdx.x; c < cols; c += 256) { float x = v[c]; s += x * x; }
  red[threadIdx.x] = s;
  __syncthreads();
  for (int off = 128; off > 0; off >>= 1) {
    if (threadIdx.x < off) red[threadIdx.x] += red[threadIdx.x + off];
    __syncthreads();
  }
  const float nrm = sqrtf(red[0]);
  const float scl = (G ? G[r] : 1.0f) / fmaxf(nrm, 1e-12f);
  __bf16* w = W + (long long)r * cols;
  for (int c = threadIdx.x; c < cols; c += 256) w[c] = f2bf(v[c] * scl);
}

// ---------------------------------------------------------------------------
// 2) batched GEMM:  Out[b] = Wbf(MxK bf16) @ Xf32[b](KxN) + bias   (f32 out)
//    128x64 tile / workgroup, 8 waves, WMMA f32_16x16x32_bf16,
//    software-pipelined global->reg->LDS staging
// ---------------------------------------------------------------------------
#define GM 128
#define GN 64
#define GK 32
#define SWS 40   // LDS row stride (elements); 80B => 16B-aligned chunks

__global__ __launch_bounds__(256) void msvq_wngemm_kernel(
    const __bf16* __restrict__ Wbf, const float* __restrict__ X,
    const float* __restrict__ bias, float* __restrict__ Out,
    int K, int ldx, int ldo, long long xBatch, long long oBatch) {
  __shared__ __bf16 sW[GM * SWS];    // [m][k]
  __shared__ __bf16 sXT[GN * SWS];   // [n][k]  (k-transposed)

  const int b  = blockIdx.z;
  const int n0 = blockIdx.x * GN;
  const int m0 = blockIdx.y * GM;
  const float* Xb = X + (long long)b * xBatch;
  float* Ob       = Out + (long long)b * oBatch;

  const int tid  = threadIdx.x;
  const int lane = tid & 31;
  const int wv   = tid >> 5;
  const int half = lane >> 4;
  const int l15  = lane & 15;

  // staging assignments
  const int mW  = tid >> 1;            // W: row, 16 contiguous k per thread
  const int ksW = (tid & 1) * 16;
  const int nX  = tid & 63;            // X: fixed n, 8 consecutive k (strided rows)
  const int k0X = (tid >> 6) * 8;

  v8f acc[4] = {v8f{}, v8f{}, v8f{}, v8f{}};

  v8bf wreg0, wreg1;
  float xreg[8];

  // prologue: prefetch k-slice 0
  {
    const __bf16* src = Wbf + (long long)(m0 + mW) * K + ksW;
    wreg0 = *reinterpret_cast<const v8bf*>(src);
    wreg1 = *reinterpret_cast<const v8bf*>(src + 8);
    const float* xs = Xb + (long long)k0X * ldx + n0 + nX;
#pragma unroll
    for (int j = 0; j < 8; j++) xreg[j] = xs[(long long)j * ldx];
  }

  for (int kb = 0; kb < K; kb += GK) {
    // commit staged slice to LDS (one b128 store for the X transpose)
    *reinterpret_cast<v8bf*>(&sW[mW * SWS + ksW])     = wreg0;
    *reinterpret_cast<v8bf*>(&sW[mW * SWS + ksW + 8]) = wreg1;
    {
      v8bf xp;
#pragma unroll
      for (int j = 0; j < 8; j++) xp[j] = f2bf(xreg[j]);
      *reinterpret_cast<v8bf*>(&sXT[nX * SWS + k0X]) = xp;
    }
    __syncthreads();

    // prefetch next k-slice (overlaps with WMMA compute below)
    if (kb + GK < K) {
      const __bf16* src = Wbf + (long long)(m0 + mW) * K + (kb + GK) + ksW;
      wreg0 = *reinterpret_cast<const v8bf*>(src);
      wreg1 = *reinterpret_cast<const v8bf*>(src + 8);
      const float* xs = Xb + (long long)(kb + GK + k0X) * ldx + n0 + nX;
#pragma unroll
      for (int j = 0; j < 8; j++) xreg[j] = xs[(long long)j * ldx];
    }

    // A fragment (16x32): lanes 0-15 K 0..7/16..23, lanes 16-31 +8
    const int mrow = wv * 16 + l15;
    v16bf afrag = cat16(*reinterpret_cast<const v8bf*>(&sW[mrow * SWS + 8 * half]),
                        *reinterpret_cast<const v8bf*>(&sW[mrow * SWS + 16 + 8 * half]));

    // load all 4 B fragments first, then issue WMMAs back-to-back
    v16bf bfrag[4];
#pragma unroll
    for (int ns = 0; ns < 4; ns++) {
      const int n = ns * 16 + l15;   // B fragment (32x16): k = 8*half + e / 16 + 8*half + (e-8)
      bfrag[ns] = cat16(*reinterpret_cast<const v8bf*>(&sXT[n * SWS + 8 * half]),
                        *reinterpret_cast<const v8bf*>(&sXT[n * SWS + 16 + 8 * half]));
    }
#pragma unroll
    for (int ns = 0; ns < 4; ns++) {
      acc[ns] = __builtin_amdgcn_wmma_f32_16x16x32_bf16(
          false, afrag, false, bfrag[ns], (short)0, acc[ns], false, false);
    }
    __syncthreads();
  }

  // epilogue: C/D layout M = j + 8*half, N = l15
  const int mbase = m0 + wv * 16 + 8 * half;
#pragma unroll
  for (int ns = 0; ns < 4; ns++) {
    const int n = n0 + ns * 16 + l15;
#pragma unroll
    for (int j = 0; j < 8; j++) {
      const int m = mbase + j;
      Ob[(long long)m * ldo + n] = acc[ns][j] + bias[m];
    }
  }
}

// ---------------------------------------------------------------------------
// 3) area downsample x4  ->  bf16 enc [(B*SC) x CB_DIM]
// ---------------------------------------------------------------------------
__global__ void msvq_pool_kernel(const float* __restrict__ z_e,
                                 __bf16* __restrict__ enc) {
  const long long i = (long long)blockIdx.x * blockDim.x + threadIdx.x;
  if (i >= (long long)BQ * CB_DIM * SC) return;
  const int s = (int)(i & (SC - 1));
  const int d = (int)((i >> 10) & (CB_DIM - 1));
  const int b = (int)(i >> 18);
  const float* p = z_e + ((long long)b * CB_DIM + d) * TT + RATIO * s;
  enc[((long long)b * SC + s) * CB_DIM + d] =
      f2bf(0.25f * (p[0] + p[1] + p[2] + p[3]));
}

// ---------------------------------------------------------------------------
// 4) similarity GEMM + argmax:  argmax_n  enc[row,:] . cn[n,:]
//    one 16-row tile per workgroup; K=256; N=8192 swept by 8 waves;
//    two independent WMMA accumulation chains per chunk
// ---------------------------------------------------------------------------
#define SAS 264  // LDS A row stride (mult of 8 -> 16B aligned chunks)

__global__ __launch_bounds__(256) void msvq_argmax_kernel(
    const __bf16* __restrict__ Enc, const __bf16* __restrict__ Cn,
    int* __restrict__ idx_out, float* __restrict__ idx_f_out) {
  __shared__ __bf16 sA[16 * SAS];
  __shared__ float sScore[16 * 128];
  __shared__ int   sIdx[16 * 128];

  const int tid  = threadIdx.x;
  const int row0 = blockIdx.x * 16;

  {  // stage 16x256 A tile
    const int m = tid >> 4, ks = (tid & 15) * 16;
    const __bf16* src = Enc + (long long)(row0 + m) * CB_DIM + ks;
    *reinterpret_cast<v8bf*>(&sA[m * SAS + ks])     = *reinterpret_cast<const v8bf*>(src);
    *reinterpret_cast<v8bf*>(&sA[m * SAS + ks + 8]) = *reinterpret_cast<const v8bf*>(src + 8);
  }
  __syncthreads();

  const int lane = tid & 31, wv = tid >> 5;
  const int half = lane >> 4, l15 = lane & 15;

  v16bf afrag[8];  // preload all K fragments for this lane's row
#pragma unroll
  for (int kk = 0; kk < 8; kk++) {
    const int off = l15 * SAS + kk * 32 + 8 * half;
    afrag[kk] = cat16(*reinterpret_cast<const v8bf*>(&sA[off]),
                      *reinterpret_cast<const v8bf*>(&sA[off + 16]));
  }

  float best[8];
  int   bestn[8];
#pragma unroll
  for (int j = 0; j < 8; j++) { best[j] = -3.0e38f; bestn[j] = 0; }

  for (int it = 0; it < CB_SIZE / 128; it++) {
    const int nbase = it * 128 + wv * 16;
    const __bf16* cb = Cn + (long long)(nbase + l15) * CB_DIM + 8 * half;
    v8f acc0 = {}, acc1 = {};
#pragma unroll
    for (int kk = 0; kk < 8; kk += 2) {
      v16bf b0 = cat16(*reinterpret_cast<const v8bf*>(cb + kk * 32),
                       *reinterpret_cast<const v8bf*>(cb + kk * 32 + 16));
      v16bf b1 = cat16(*reinterpret_cast<const v8bf*>(cb + kk * 32 + 32),
                       *reinterpret_cast<const v8bf*>(cb + kk * 32 + 48));
      acc0 = __builtin_amdgcn_wmma_f32_16x16x32_bf16(
          false, afrag[kk], false, b0, (short)0, acc0, false, false);
      acc1 = __builtin_amdgcn_wmma_f32_16x16x32_bf16(
          false, afrag[kk + 1], false, b1, (short)0, acc1, false, false);
    }
    const v8f acc = acc0 + acc1;
    const int n = nbase + l15;
#pragma unroll
    for (int j = 0; j < 8; j++)   // strict > keeps smallest n within subset
      if (acc[j] > best[j]) { best[j] = acc[j]; bestn[j] = n; }
  }

#pragma unroll
  for (int j = 0; j < 8; j++) {
    const int r = j + 8 * half;
    sScore[r * 128 + wv * 16 + l15] = best[j];
    sIdx[r * 128 + wv * 16 + l15]   = bestn[j];
  }
  __syncthreads();
  if (tid < 16) {
    float bs = -3.0e38f; int bi = 0x7fffffff;
    for (int c = 0; c < 128; c++) {
      const float s = sScore[tid * 128 + c];
      const int   n = sIdx[tid * 128 + c];
      if (s > bs || (s == bs && n < bi)) { bs = s; bi = n; }
    }
    idx_out[row0 + tid]   = bi;
    idx_f_out[row0 + tid] = (float)bi;
  }
}

// ---------------------------------------------------------------------------
// 5) gather codebook rows -> z_q [b][s][d]  (f32)
// ---------------------------------------------------------------------------
__global__ void msvq_gather_kernel(const float* __restrict__ codebook,
                                   const int* __restrict__ idx,
                                   float* __restrict__ zq) {
  const long long i = (long long)blockIdx.x * blockDim.x + threadIdx.x;
  if (i >= (long long)BQ * SC * CB_DIM) return;
  const int d = (int)(i & (CB_DIM - 1));
  const long long bs = i >> 8;
  zq[i] = codebook[(long long)idx[bs] * CB_DIM + d];
}

// ---------------------------------------------------------------------------
// 6) linear upsample (align_corners=False) + fused MSE loss accumulation
// ---------------------------------------------------------------------------
__global__ __launch_bounds__(256) void msvq_upsample_loss_kernel(
    const float* __restrict__ zq,    // [b][s][d]
    const float* __restrict__ z_e,   // [b][d][T]
    float* __restrict__ zq_up,       // [b][d][T]
    float* __restrict__ loss_acc) {  // [B]
  __shared__ float red[256];
  const int b = blockIdx.y;
  const long long base = (long long)b * CB_DIM * TT;
  const float* zb = zq + (long long)b * SC * CB_DIM;
  float lsum = 0.f;
#pragma unroll 1
  for (int it = 0; it < 16; it++) {
    const long long i = (long long)blockIdx.x * 4096 + it * 256 + threadIdx.x;
    const int d = (int)(i >> 12);
    const int t = (int)(i & (TT - 1));
    const float src = fmaxf((t + 0.5f) * 0.25f - 0.5f, 0.0f);
    const int i0 = (int)src;                    // src >= 0, trunc == floor
    const float frac = src - (float)i0;
    const int i1 = min(i0 + 1, SC - 1);
    const float v0 = zb[(long long)i0 * CB_DIM + d];
    const float v1 = zb[(long long)i1 * CB_DIM + d];
    const float v = v0 + (v1 - v0) * frac;
    zq_up[base + i] = v;
    const float diff = z_e[base + i] - v;
    lsum += diff * diff;
  }
  red[threadIdx.x] = lsum;
  __syncthreads();
  for (int off = 128; off > 0; off >>= 1) {
    if (threadIdx.x < off) red[threadIdx.x] += red[threadIdx.x + off];
    __syncthreads();
  }
  if (threadIdx.x == 0) atomicAdd(&loss_acc[b], red[0]);
}

// ---------------------------------------------------------------------------
// 7) finalize losses (commitment == codebook loss forward value)
// ---------------------------------------------------------------------------
__global__ void msvq_finalize_kernel(const float* __restrict__ loss_acc,
                                     float* __restrict__ out_commit,
                                     float* __restrict__ out_cb) {
  const int b = threadIdx.x;
  if (b < BQ) {
    const float v = loss_acc[b] * (1.0f / ((float)CB_DIM * (float)TT));
    out_commit[b] = v;
    out_cb[b]     = v;
  }
}

// ---------------------------------------------------------------------------
// host launcher
// ---------------------------------------------------------------------------
extern "C" void kernel_launch(void* const* d_in, const int* in_sizes, int n_in,
                              void* d_out, int out_size, void* d_ws, size_t ws_size,
                              hipStream_t stream) {
  (void)in_sizes; (void)n_in; (void)out_size; (void)ws_size;

  const float* z        = (const float*)d_in[0];
  const float* inp_v    = (const float*)d_in[1];
  const float* inp_g    = (const float*)d_in[2];
  const float* inp_b    = (const float*)d_in[3];
  const float* outp_v   = (const float*)d_in[4];
  const float* outp_g   = (const float*)d_in[5];
  const float* outp_b   = (const float*)d_in[6];
  const float* codebook = (const float*)d_in[7];

  float* out = (float*)d_out;
  // output layout (flattened tuple, f32)
  float* out_zqout  = out;                                    // B*D_IN*T
  float* out_commit = out + (long long)BQ * D_IN * TT;        // B
  float* out_cb     = out_commit + BQ;                        // B
  float* out_idx    = out_cb + BQ;                            // B*SC
  float* out_ze     = out_idx + (long long)BQ * SC;           // B*CB_DIM*T

  // workspace carve-up
  char* ws = (char*)d_ws;
  __bf16* w_in_bf  = (__bf16*)ws;                 ws += (long long)CB_DIM * D_IN * 2;
  __bf16* w_out_bf = (__bf16*)ws;                 ws += (long long)D_IN * CB_DIM * 2;
  __bf16* cn_bf    = (__bf16*)ws;                 ws += (long long)CB_SIZE * CB_DIM * 2;
  __bf16* enc_bf   = (__bf16*)ws;                 ws += (long long)BQ * SC * CB_DIM * 2;
  int*    idx_ws   = (int*)ws;                    ws += (long long)BQ * SC * 4;
  float*  zq_ws    = (float*)ws;                  ws += (long long)BQ * SC * CB_DIM * 4;
  float*  zqup_ws  = (float*)ws;                  ws += (long long)BQ * CB_DIM * TT * 4;
  float*  loss_acc = (float*)ws;

  msvq_zero_kernel<<<1, 64, 0, stream>>>(loss_acc, BQ);

  // weight-norm weights + normalized codebook -> bf16
  msvq_rownorm_kernel<<<CB_DIM, 256, 0, stream>>>(inp_v, inp_g, w_in_bf, D_IN);
  msvq_rownorm_kernel<<<D_IN, 256, 0, stream>>>(outp_v, outp_g, w_out_bf, CB_DIM);
  msvq_rownorm_kernel<<<CB_SIZE, 256, 0, stream>>>(codebook, nullptr, cn_bf, CB_DIM);

  // z_e = W_in @ z + b   -> written directly into the z_e output slot
  {
    dim3 grid(TT / GN, CB_DIM / GM, BQ);
    msvq_wngemm_kernel<<<grid, 256, 0, stream>>>(
        w_in_bf, z, inp_b, out_ze,
        D_IN, TT, TT, (long long)D_IN * TT, (long long)CB_DIM * TT);
  }

  // area downsample -> bf16 enc
  msvq_pool_kernel<<<(BQ * CB_DIM * SC) / 256, 256, 0, stream>>>(out_ze, enc_bf);

  // cosine-sim GEMM + argmax (writes int ws + float output indices)
  msvq_argmax_kernel<<<(BQ * SC) / 16, 256, 0, stream>>>(enc_bf, cn_bf, idx_ws, out_idx);

  // gather, upsample + losses
  msvq_gather_kernel<<<(BQ * SC * CB_DIM) / 256, 256, 0, stream>>>(codebook, idx_ws, zq_ws);
  {
    dim3 grid((CB_DIM * TT) / 4096, BQ);
    msvq_upsample_loss_kernel<<<grid, 256, 0, stream>>>(zq_ws, out_ze, zqup_ws, loss_acc);
  }

  // z_q_out = W_out @ zq_up + b
  {
    dim3 grid(TT / GN, D_IN / GM, BQ);
    msvq_wngemm_kernel<<<grid, 256, 0, stream>>>(
        w_out_bf, zqup_ws, outp_b, out_zqout,
        CB_DIM, TT, TT, (long long)CB_DIM * TT, (long long)D_IN * TT);
  }

  msvq_finalize_kernel<<<1, 64, 0, stream>>>(loss_acc, out_commit, out_cb);
}
